// DepthAugmentedBEVLifter_82703890252457
// MI455X (gfx1250) — matrix operand order repaired and checked by reading, not compiled, not run on hardware
//
#include <hip/hip_runtime.h>
#include <hip/hip_bf16.h>

// ---------------------------------------------------------------------------
// CDNA5 (gfx1250) WMMA-based depth-augmented BEV lifter.
// wave32; channel contractions on v_wmma_f32_16x16x32_f16 (f32 accum, f16
// operands staged through LDS in fragment-native layouts -> ds_load_b128),
// weight tiles staged via GLOBAL_LOAD_ASYNC_TO_LDS (ASYNCcnt path).
// ---------------------------------------------------------------------------

typedef __attribute__((ext_vector_type(16))) _Float16 v16h;
typedef __attribute__((ext_vector_type(8)))  _Float16 v8h;
typedef __attribute__((ext_vector_type(4)))  _Float16 v4h;
typedef __attribute__((ext_vector_type(2)))  _Float16 v2h;
typedef __attribute__((ext_vector_type(8)))  float    v8f;

#define EPI_BN_RELU      0
#define EPI_BIAS         1
#define EPI_BIAS_SIGMOID 2

#define BEV_HW 16384   // 128*128
#define NB     4       // batch

__device__ __forceinline__ v16h cat8(v8h lo, v8h hi) {
  return __builtin_shufflevector(lo, hi, 0,1,2,3,4,5,6,7,8,9,10,11,12,13,14,15);
}

// Generic LDS pointer -> 32-bit LDS address (ISA §10.2: LDS aperture addresses
// truncate to LDS_ADDR in the low 32 bits).
__device__ __forceinline__ unsigned lds_addr32(const void* p) {
  return (unsigned)(unsigned long long)p;
}

// A tile (MxK) row-major in LDS, row stride `ld` halves (row byte-stride
// multiple of 16). Lane (m=lane&15, h=lane>>4) needs K = h*8+{0..7} and
// 16+h*8+{0..7} -> two aligned 16B loads.
__device__ __forceinline__ v16h frag_a(const _Float16* __restrict__ t, int ld) {
  int lane = threadIdx.x & 31;
  const _Float16* base = t + (lane & 15) * ld + (lane >> 4) * 8;
  return cat8(*(const v8h*)(base), *(const v8h*)(base + 16));
}

// B tile (KxN) column-major in LDS (one column = one pixel's K values),
// column stride `ld` halves (byte-stride multiple of 16).
// Lane (n=lane&15, h=lane>>4) needs K = h*8+{0..7} and 16+h*8+{0..7} -> two aligned 16B loads.
__device__ __forceinline__ v16h frag_b(const _Float16* __restrict__ t, int ld) {
  int lane = threadIdx.x & 31;
  const _Float16* base = t + (lane & 15) * ld + (lane >> 4) * 8;
  return cat8(*(const v8h*)(base), *(const v8h*)(base + 16));
}

// ---- 1x1 conv as GEMM: Y(B,Cout,HW) = epi( W(Cout,Cin) @ X(B,Cin,HW) ) -----
// 128 threads = 4 waves. Tile M=64 (padded), N=64 (16/wave), K step 64
// (2 slabs -> 8 wmma per iteration). Strides padded to 72 halves (bank-spread
// permutation 36n mod 64, 16B-aligned). Weight tiles come in through the
// async-to-LDS pipe (f32), then get converted into the f16 fragment layout.
__global__ __launch_bounds__(128) void k_gemm1x1(
    const float* __restrict__ X, const float* __restrict__ Wt,
    const float* __restrict__ g, const float* __restrict__ bt,
    float* __restrict__ Y, int Cin, int Cout, int HW, int mode)
{
  __shared__ __align__(16) _Float16 sA[64 * 72];   // f16 weights, row-major, stride 72
  __shared__ __align__(16) _Float16 sB[64 * 72];   // f16 acts, column-major, stride 72
  __shared__ __align__(16) float    sW32[64 * 64]; // async f32 weight staging
  const int b    = blockIdx.z;
  const int m0   = blockIdx.y * 64;
  const int p0   = blockIdx.x * 64;
  const int tid  = threadIdx.x;
  const int wave = tid >> 5;
  const float* __restrict__ Xb = X + (size_t)b * Cin * HW;
  const bool interior = (p0 + 64 <= HW);
  const bool fullM    = (m0 + 64 <= Cout);

  v8f acc[4] = {};
  for (int k0 = 0; k0 < Cin; k0 += 64) {
    // ---- stage weights (64 rows x 64 K) -> sA (f16, fragment layout)
    if (fullM) {
      // CDNA5 async DMA: 16B per lane, global -> LDS, tracked by ASYNCcnt
#pragma unroll
      for (int i = 0; i < 8; ++i) {
        int c  = tid + i * 128;          // 16B chunk id, 0..1023
        int m  = c >> 4;
        int k4 = (c & 15) * 4;
        const float* src = &Wt[(size_t)(m0 + m) * Cin + k0 + k4];
        unsigned dst = lds_addr32(&sW32[m * 64 + k4]);
        asm volatile("global_load_async_to_lds_b128 %0, %1, off"
                     :: "v"(dst), "v"(src) : "memory");
      }
      asm volatile("s_wait_asynccnt 0" ::: "memory");
      __syncthreads();                   // publish async-written LDS
#pragma unroll
      for (int i = 0; i < 8; ++i) {
        int c  = tid + i * 128;
        int m  = c >> 4;
        int k4 = (c & 15) * 4;
        float4 wv = *(const float4*)&sW32[m * 64 + k4];
        v4h h = { (_Float16)wv.x, (_Float16)wv.y, (_Float16)wv.z, (_Float16)wv.w };
        *(v4h*)&sA[m * 72 + k4] = h;
      }
    } else {
      int m  = tid >> 1;
      int kc = (tid & 1) * 32;
      bool valid = (m0 + m < Cout);
      const float* wrow = &Wt[(size_t)(m0 + m) * Cin + k0 + kc];
#pragma unroll
      for (int j = 0; j < 8; ++j) {
        float4 wv = valid ? *(const float4*)(wrow + j * 4)
                          : make_float4(0.f, 0.f, 0.f, 0.f);
        v4h h = { (_Float16)wv.x, (_Float16)wv.y, (_Float16)wv.z, (_Float16)wv.w };
        *(v4h*)&sA[m * 72 + kc + j * 4] = h;
      }
    }
    // ---- stage activations 64K x 64px into column-major sB
    if (interior) {
#pragma unroll
      for (int i = 0; i < 4; ++i) {
        int c  = tid + i * 128;        // 0..511 chunks of (2 rows x 4 cols)
        int kp = c >> 4;               // row pair 0..31
        int nc = (c & 15) * 4;         // column base
        const float* src = &Xb[(size_t)(k0 + 2 * kp) * HW + p0 + nc];
        float4 r0 = *(const float4*)(src);
        float4 r1 = *(const float4*)(src + HW);
        if (k0 + 64 < Cin) __builtin_prefetch(src + (size_t)64 * HW, 0, 1);
        v2h c0 = { (_Float16)r0.x, (_Float16)r1.x };
        v2h c1 = { (_Float16)r0.y, (_Float16)r1.y };
        v2h c2 = { (_Float16)r0.z, (_Float16)r1.z };
        v2h c3 = { (_Float16)r0.w, (_Float16)r1.w };
        *(v2h*)&sB[(nc + 0) * 72 + 2 * kp] = c0;
        *(v2h*)&sB[(nc + 1) * 72 + 2 * kp] = c1;
        *(v2h*)&sB[(nc + 2) * 72 + 2 * kp] = c2;
        *(v2h*)&sB[(nc + 3) * 72 + 2 * kp] = c3;
      }
    } else {
#pragma unroll
      for (int i = 0; i < 32; ++i) {
        int e = tid + i * 128;         // 0..4095
        int k = e >> 6, n = e & 63;
        int p = p0 + n;
        float xv = (p < HW) ? Xb[(size_t)(k0 + k) * HW + p] : 0.f;
        sB[n * 72 + k] = (_Float16)xv;
      }
    }
    __syncthreads();
#pragma unroll
    for (int s = 0; s < 2; ++s) {      // two K=32 slabs
      v16h fb = frag_b(&sB[(wave * 16) * 72 + s * 32], 72);
#pragma unroll
      for (int ms = 0; ms < 4; ++ms) {
        v16h fa = frag_a(&sA[(ms * 16) * 72 + s * 32], 72);
        acc[ms] = __builtin_amdgcn_wmma_f32_16x16x32_f16(
            false, fa, false, fb, (short)0, acc[ms], false, false);
      }
    }
    __syncthreads();
  }
  // epilogue (C layout: n = lane&15, m = vgpr + (lane>>4)*8)
  const int lane = tid & 31;
  const int n = lane & 15, half = lane >> 4;
  const int p = p0 + wave * 16 + n;
  if (p >= HW) return;
#pragma unroll
  for (int ms = 0; ms < 4; ++ms) {
#pragma unroll
    for (int v = 0; v < 8; ++v) {
      int m = m0 + ms * 16 + half * 8 + v;
      if (m >= Cout) continue;
      float val = acc[ms][v];
      float o;
      if (mode == EPI_BN_RELU)      o = fmaxf(val * g[m] + bt[m], 0.f);
      else if (mode == EPI_BIAS)    o = val + g[m];
      else                          o = 1.f / (1.f + __expf(-(val + g[m])));
      Y[((size_t)b * Cout + m) * HW + p] = o;
    }
  }
}

// ---- 3x3 conv (stride 1, pad 1) as implicit GEMM, grouped + concat input ---
// Input channel c<C0 from X0 else X1 (channel c-C0). K = Cin_pg*9 zero-padded
// to multiple of 32. 256 threads = 8 waves; tile M=32 (2 wmma/step), N=128.
// BN+ReLU. LDS strides padded to 40 halves (bank permutation 20n mod 64).
__global__ __launch_bounds__(256) void k_conv3x3(
    const float* __restrict__ X0, int C0,
    const float* __restrict__ X1, int C1,
    const float* __restrict__ Wt,  // (groups*Cout_pg, Cin_pg, 3, 3)
    const float* __restrict__ g, const float* __restrict__ bt,
    float* __restrict__ Y,
    int Cin_pg, int Cout_pg, int groups, int H, int W)
{
  __shared__ __align__(16) _Float16 sA[32 * 40];    // row-major, stride 40
  __shared__ __align__(16) _Float16 sB[128 * 40];   // column-major, stride 40
  const int HW = H * W;
  const int b  = blockIdx.z;
  const int mtiles_pg = (Cout_pg + 31) >> 5;
  const int grp = blockIdx.y / mtiles_pg;
  const int m0  = (blockIdx.y % mtiles_pg) * 32;
  const int p0  = blockIdx.x * 128;
  const int tid = threadIdx.x;
  const int wave = tid >> 5;
  const int Ktot = Cin_pg * 9;
  const int Cout_tot = groups * Cout_pg;

  v8f acc0 = {}, acc1 = {};
  for (int k0 = 0; k0 < Ktot; k0 += 32) {
    // stage weights 32x32
#pragma unroll
    for (int i = 0; i < 4; ++i) {
      int e = tid + i * 256;
      int m = e >> 5, k = e & 31;
      int gk = k0 + k;
      float wv = 0.f;
      if (m0 + m < Cout_pg && gk < Ktot) {
        int il = gk / 9, r = gk % 9;
        wv = Wt[(((size_t)(grp * Cout_pg + m0 + m) * Cin_pg + il) * 9) + r];
      }
      sA[m * 40 + k] = (_Float16)wv;
    }
    // gather activation slab 32K x 128px (im2col on the fly, zero halo)
#pragma unroll
    for (int i = 0; i < 16; ++i) {
      int e = tid + i * 256;
      int k = e >> 7, n = e & 127;
      int gk = k0 + k;
      int p = p0 + n;
      float xv = 0.f;
      if (gk < Ktot && p < HW) {
        int il = gk / 9, r = gk % 9;
        int hh = p / W + r / 3 - 1;
        int ww = p % W + r % 3 - 1;
        if (hh >= 0 && hh < H && ww >= 0 && ww < W) {
          int c = grp * Cin_pg + il;
          xv = (c < C0) ? X0[((size_t)b * C0 + c) * HW + hh * W + ww]
                        : X1[((size_t)b * C1 + (c - C0)) * HW + hh * W + ww];
        }
      }
      sB[n * 40 + k] = (_Float16)xv;
    }
    __syncthreads();
    v16h fb = frag_b(&sB[(wave * 16) * 40], 40);
    v16h fa0 = frag_a(&sA[0], 40);
    v16h fa1 = frag_a(&sA[16 * 40], 40);
    acc0 = __builtin_amdgcn_wmma_f32_16x16x32_f16(
        false, fa0, false, fb, (short)0, acc0, false, false);
    acc1 = __builtin_amdgcn_wmma_f32_16x16x32_f16(
        false, fa1, false, fb, (short)0, acc1, false, false);
    __syncthreads();
  }
  const int lane = tid & 31;
  const int n = lane & 15, half = lane >> 4;
  const int p = p0 + wave * 16 + n;
  if (p < HW) {
#pragma unroll
    for (int v = 0; v < 8; ++v) {
      int ml = m0 + half * 8 + v;
      if (ml < Cout_pg) {
        int m = grp * Cout_pg + ml;
        Y[((size_t)b * Cout_tot + m) * HW + p] =
            fmaxf(acc0[v] * g[m] + bt[m], 0.f);
      }
      int ml1 = ml + 16;
      if (ml1 < Cout_pg) {
        int m = grp * Cout_pg + ml1;
        Y[((size_t)b * Cout_tot + m) * HW + p] =
            fmaxf(acc1[v] * g[m] + bt[m], 0.f);
      }
    }
  }
}

// ---- depth softmax + projection + BEV scatter (one wave32 per pixel) -------
__global__ __launch_bounds__(256) void k_splat(
    const float* __restrict__ dl,    // (B,64,HW) depth logits
    const float* __restrict__ red,   // (B,128,HW)
    const float* __restrict__ cf,    // (B,1,HW)
    const float* __restrict__ Kmat,  // (B,3,3)
    const float* __restrict__ Tmat,  // (B,4,4)
    const float* __restrict__ bins,  // (64)
    float* __restrict__ acc,         // (B,128,128,128)
    int H, int W)
{
  const int HW = H * W;
  const int lane = threadIdx.x & 31;
  const int wid = blockIdx.x * (blockDim.x >> 5) + (threadIdx.x >> 5);
  const int nwaves = gridDim.x * (blockDim.x >> 5);
  for (int pg = wid; pg < NB * HW; pg += nwaves) {
    int b = pg / HW, p = pg % HW;
    // expected depth: softmax(dl*10) . bins, 2 bins per lane + xor reductions
    const float* dlb = dl + (size_t)b * 64 * HW + p;
    float x0 = dlb[(size_t)lane * HW] * 10.f;
    float x1 = dlb[(size_t)(lane + 32) * HW] * 10.f;
    float mx = fmaxf(x0, x1);
#pragma unroll
    for (int off = 16; off; off >>= 1) mx = fmaxf(mx, __shfl_xor(mx, off, 32));
    float e0 = __expf(x0 - mx), e1 = __expf(x1 - mx);
    float s = e0 + e1;
    float d = e0 * bins[lane] + e1 * bins[lane + 32];
#pragma unroll
    for (int off = 16; off; off >>= 1) {
      s += __shfl_xor(s, off, 32);
      d += __shfl_xor(d, off, 32);
    }
    float depth = d / s;
    // analytic 3x3 inverse of intrinsics
    const float* K = Kmat + b * 9;
    float a00=K[0],a01=K[1],a02=K[2],a10=K[3],a11=K[4],a12=K[5],a20=K[6],a21=K[7],a22=K[8];
    float id = 1.f / (a00*(a11*a22-a12*a21) - a01*(a10*a22-a12*a20) + a02*(a10*a21-a11*a20));
    float px = (float)(p % W), py = (float)(p / W);
    float cx = depth * ((a11*a22-a12*a21)*px + (a02*a21-a01*a22)*py + (a01*a12-a02*a11)) * id;
    float cy = depth * ((a12*a20-a10*a22)*px + (a00*a22-a02*a20)*py + (a02*a10-a00*a12)) * id;
    float cz = depth * ((a10*a21-a11*a20)*px + (a01*a20-a00*a21)*py + (a00*a11-a01*a10)) * id;
    const float* T = Tmat + b * 16;
    float ex = T[0]*cx + T[1]*cy + T[2]*cz + T[3];
    float ey = T[4]*cx + T[5]*cy + T[6]*cz + T[7];
    int bx = (int)(ex * 2.5f + 64.f);   // 1/VOX = 2.5, trunc toward zero
    int by = (int)(ey * 2.5f + 64.f);
    if (bx < 0 || bx >= 128 || by < 0 || by >= 128) continue;  // wave-uniform
    float cfv = cf[(size_t)b * HW + p];
    const float* redb = red + (size_t)b * 128 * HW + p;
    float* accb = acc + (size_t)b * 128 * BEV_HW + (by * 128 + bx);
#pragma unroll
    for (int i = 0; i < 4; ++i) {
      int ch = lane + i * 32;
      atomicAdd(&accb[(size_t)ch * BEV_HW], redb[(size_t)ch * HW] * cfv);
    }
  }
}

__global__ void k_zero(float* __restrict__ p, size_t n) {
  size_t i = (size_t)blockIdx.x * blockDim.x + threadIdx.x;
  if (i < n) p[i] = 0.f;
}

// ---------------------------------------------------------------------------
extern "C" void kernel_launch(void* const* d_in, const int* in_sizes, int n_in,
                              void* d_out, int out_size, void* d_ws, size_t ws_size,
                              hipStream_t stream) {
  auto F = [&](int i) { return (const float*)d_in[i]; };
  const float* feat3 = F(0);           // (4,256,96,176)
  const float* feat5 = F(1);           // (4,512,24,44)
  const float* intr  = F(2);           // (4,3,3)
  const float* extr  = F(3);           // (4,4,4)
  // params: stage3 [4..19], stage5 [20..35], skip [36..38], main [39..41]
  const float* bins  = F(42);          // (64)

  // workspace carve (sized for stage3, reused for stage5)
  const int HW3 = 96 * 176;
  float* ws = (float*)d_ws;
  size_t off = 0;
  float* red1 = ws + off; off += (size_t)NB * 64 * HW3;
  float* dl1  = ws + off; off += (size_t)NB * 64 * HW3;
  float* dlg  = ws + off; off += (size_t)NB * 64 * HW3;
  float* red  = ws + off; off += (size_t)NB * 128 * HW3;
  float* cf64 = ws + off; off += (size_t)NB * 64 * HW3;
  float* cf1  = ws + off; off += (size_t)NB * 1 * HW3;
  float* acc3 = ws + off; off += (size_t)NB * 128 * BEV_HW;
  float* acc5 = ws + off; off += (size_t)NB * 128 * BEV_HW;
  (void)ws_size; (void)n_in; (void)in_sizes; (void)out_size;

  // zero both BEV accumulators (contiguous)
  {
    size_t n = (size_t)2 * NB * 128 * BEV_HW;
    k_zero<<<(unsigned)((n + 255) / 256), 256, 0, stream>>>(acc3, n);
  }

  auto gemm = [&](const float* X, const float* Wt, const float* g,
                  const float* bt, float* Y, int Cin, int Cout, int HW, int mode) {
    dim3 gr((HW + 63) / 64, (Cout + 63) / 64, NB);
    k_gemm1x1<<<gr, 128, 0, stream>>>(X, Wt, g, bt, Y, Cin, Cout, HW, mode);
  };
  auto conv3 = [&](const float* X0, int C0, const float* X1, int C1,
                   const float* Wt, const float* g, const float* bt, float* Y,
                   int Cin_pg, int Cout_pg, int groups, int H, int W) {
    int HW = H * W;
    dim3 gr((HW + 127) / 128, groups * ((Cout_pg + 31) / 32), NB);
    k_conv3x3<<<gr, 256, 0, stream>>>(X0, C0, X1, C1, Wt, g, bt, Y,
                                      Cin_pg, Cout_pg, groups, H, W);
  };

  auto lift = [&](const float* feat, int C, int H, int W, int pb, float* accum) {
    const float *rw1=F(pb+0), *rg1=F(pb+1), *rb1=F(pb+2), *rw2=F(pb+3),
                *rg2=F(pb+4), *rb2=F(pb+5), *dw1=F(pb+6), *dg1=F(pb+7),
                *db1=F(pb+8), *dw2=F(pb+9), *dwb2=F(pb+10), *cw1=F(pb+11),
                *cg1=F(pb+12), *cb1=F(pb+13), *cw2=F(pb+14), *cwb2=F(pb+15);
    int HW = H * W;
    gemm(feat, rw1, rg1, rb1, red1, C, 64, HW, EPI_BN_RELU);          // reduce 1x1
    gemm(feat, dw1, dg1, db1, dl1,  C, 64, HW, EPI_BN_RELU);          // depth 1x1
    gemm(dl1,  dw2, dwb2, nullptr, dlg, 64, 64, HW, EPI_BIAS);        // depth logits
    conv3(red1, 64, red1, 64, rw2, rg2, rb2, red, 8, 16, 8, H, W);    // grouped 3x3
    conv3(dlg, 64, red, 128, cw1, cg1, cb1, cf64, 192, 64, 1, H, W);  // conf 3x3 on concat
    gemm(cf64, cw2, cwb2, nullptr, cf1, 64, 1, HW, EPI_BIAS_SIGMOID); // conf 1x1 + sigmoid
    int nblk = (NB * HW + 7) / 8;                                     // 8 waves/block
    k_splat<<<nblk, 256, 0, stream>>>(dlg, red, cf1, intr, extr, bins, accum, H, W);
  };

  lift(feat3, 256, 96, 176, 4,  acc3);
  lift(feat5, 512, 24, 44,  20, acc5);

  // outputs: (main, skip) concatenated; each (4,128,128,128)
  float* out_main = (float*)d_out;
  float* out_skip = out_main + (size_t)NB * 128 * BEV_HW;
  conv3(acc5, 128, acc5, 128, F(39), F(40), F(41), out_main, 128, 128, 1, 128, 128);
  gemm(acc3, F(36), F(37), F(38), out_skip, 128, 128, BEV_HW, EPI_BN_RELU);
}
